// InfoNCECauchy_32074815767014
// MI455X (gfx1250) — compile-verified, matrix operand order checked by reading, not run on hardware
//
#include <hip/hip_runtime.h>
#include <math.h>

// InfoNCE-Cauchy loss for MI455X (gfx1250, wave32, WMMA + TDM).
// Never materialize the 8192x8192 similarity matrices; compute 128x128 tiles
// with v_wmma_f32_16x16x32_bf16 (fp32 emulated as bf16 hi/lo, 3 products) and
// fold each tile into row/col partial sums immediately.
// Staging uses the Tensor Data Mover: one wave issues tensor_load_to_lds for
// the next k-stage (double-buffered LDS) while all waves run WMMAs on the
// current stage -- no VGPR round trip, async under TENSORcnt.
// All reductions are fixed-order (no float atomics) => deterministic.

typedef __bf16 bf16_t;
typedef __attribute__((ext_vector_type(16))) __bf16 v16bf;
typedef __attribute__((ext_vector_type(8)))  float  v8f;

typedef unsigned int u32x4 __attribute__((ext_vector_type(4)));
typedef int          i32x8 __attribute__((ext_vector_type(8)));
typedef int          i32x4 __attribute__((ext_vector_type(4)));

#define NTOT  16384
#define NHALF 8192
#define KDIM  256
#define BM    128
#define BN    128
#define BK    32
#define LDST  40     // LDS row stride in bf16 elems = 80B (64B data + 16B TDM pad)
#define TILES 64     // NHALF / BM
#define KSTAGES (KDIM / BK)
#define TILE_ELEMS (BM * LDST)   // one staged tile (padded) in bf16 elems

union FragU {
  uint4 q[2];
  v16bf v;
};

// ---------------------------------------------------------------------------
// TDM: DMA one 128x32 bf16 tile (row stride 256 elems) into LDS, inserting
// 16B of padding after every 64B row => LDS row stride 80B (= LDST elems).
// D# per CDNA5 ISA (group0 128b, group1 256b, 2D tensor).
// This toolchain exposes the 6-arg builtin form (extra int32x8, zero-filled).
// ---------------------------------------------------------------------------
__device__ __forceinline__ void tdm_load_tile(const bf16_t* gtile, unsigned lds_addr) {
  const unsigned long long ga = (unsigned long long)(uintptr_t)gtile;
  const u32x4 g0 = {
      1u,                                           // count=1 valid descriptor
      lds_addr,                                     // LDS byte address
      (unsigned)(ga & 0xffffffffu),                 // global addr [31:0]
      (unsigned)(ga >> 32) | 0x80000000u            // global addr [56:32] | type=2
  };
  const i32x8 g1 = {
      (int)((1u << 16)      // data_size = 1 -> 2 bytes (bf16)
          | (1u << 20)      // pad_enable
          | (3u << 22)      // pad_interval = 3 -> every 16 DWORDs (64B row)
          | (3u << 25)),    // pad_amount   = 3 -> 4 DWORDs (16B)
      (int)(((unsigned)KDIM & 0xffffu) << 16),      // tensor_dim0 lo16 (=256)
      (int)(((unsigned)NHALF & 0xffffu) << 16),     // dim0 hi16=0 | tensor_dim1 lo16 (=8192)
      (int)(32u << 16),                             // dim1 hi16=0 | tile_dim0 = 32 elems
      128,                                          // tile_dim1 = 128 rows | tile_dim2 = 0
      KDIM,                                         // tensor_dim0_stride lo32 (=256 elems)
      0, 0
  };
  const i32x4 z4 = {0, 0, 0, 0};                    // 2D tensor: groups 2/3 unused
  const i32x8 z8 = {0, 0, 0, 0, 0, 0, 0, 0};        // extra group (6-arg form), unused
  __builtin_amdgcn_tensor_load_to_lds(g0, g1, z4, z4, z8, 0);
}

// ---------------------------------------------------------------------------
// 1) fp32 -> bf16 hi/lo split + exact fp32 squared row norms
// ---------------------------------------------------------------------------
__global__ void prep_kernel(const float* __restrict__ f,
                            bf16_t* __restrict__ hi, bf16_t* __restrict__ lo,
                            float* __restrict__ norms) {
  const int row = blockIdx.x;      // 16384 rows
  const int t   = threadIdx.x;     // 256 threads == KDIM
  const size_t idx = (size_t)row * KDIM + t;
  float x  = f[idx];
  bf16_t h = (bf16_t)x;
  bf16_t l = (bf16_t)(x - (float)h);
  hi[idx] = h;
  lo[idx] = l;

  __shared__ float red[KDIM];
  red[t] = x * x;
  __syncthreads();
  for (int s = KDIM / 2; s > 0; s >>= 1) {
    if (t < s) red[t] += red[t + s];
    __syncthreads();
  }
  if (t == 0) norms[row] = red[0];
}

// ---------------------------------------------------------------------------
// 2) Tiled Cauchy-similarity "GEMM" with fused row/col-sum + diag extraction.
//    Workgroup = 256 threads = 8 wave32; each wave owns 16 rows x 128 cols.
//    rsPart[blockIdx.x][row]  : per-column-tile partial row sums (unique slot)
//    csPart[blockIdx.y][col]  : per-row-tile partial col sums   (unique slot)
// ---------------------------------------------------------------------------
template <bool WANT_COL, bool WANT_DIAG, bool EXCL_DIAG>
__global__ __launch_bounds__(256)
void sim_tile_kernel(const bf16_t* __restrict__ Xhi, const bf16_t* __restrict__ Xlo,
                     const float* __restrict__ nX,
                     const bf16_t* __restrict__ Yhi, const bf16_t* __restrict__ Ylo,
                     const float* __restrict__ nY,
                     float* __restrict__ rsPart, float* __restrict__ csPart,
                     float* __restrict__ diag) {
  // Double-buffered staged tiles (TDM destination), padded to LDST stride.
  __shared__ bf16_t sAhi[2 * TILE_ELEMS];
  __shared__ bf16_t sAlo[2 * TILE_ELEMS];
  __shared__ bf16_t sBhi[2 * TILE_ELEMS];
  __shared__ bf16_t sBlo[2 * TILE_ELEMS];
  __shared__ float  sCol[8 * BN];

  const int tid     = threadIdx.x;
  const int wave    = tid >> 5;
  const int lane    = tid & 31;
  const int h       = lane >> 4;   // half-wave select (K sub-block per ISA layout)
  const int l15     = lane & 15;
  const int rowBase = blockIdx.y * BM;
  const int colBase = blockIdx.x * BN;

  const size_t gArow = (size_t)rowBase * KDIM;   // A tile: rows rowBase..+127
  const size_t gBrow = (size_t)colBase * KDIM;   // B tile: rows colBase..+127

  v8f acc[8] = {};

  // Kick off stage 0 DMA (one wave issues; EXEC-independent, per-wave op).
  if (wave == 0) {
    tdm_load_tile(&Xhi[gArow], (unsigned)(uintptr_t)&sAhi[0]);
    tdm_load_tile(&Xlo[gArow], (unsigned)(uintptr_t)&sAlo[0]);
    tdm_load_tile(&Yhi[gBrow], (unsigned)(uintptr_t)&sBhi[0]);
    tdm_load_tile(&Ylo[gBrow], (unsigned)(uintptr_t)&sBlo[0]);
  }

  int buf = 0;
  for (int ks = 0; ks < KSTAGES; ++ks) {
    if (wave == 0) {
      __builtin_amdgcn_s_wait_tensorcnt(0);   // stage ks landed in LDS
    }
    __syncthreads();                          // visible to all waves

    // Async DMA of stage ks+1 into the other buffer, overlapping the WMMAs.
    if (wave == 0 && ks + 1 < KSTAGES) {
      const int k0n = (ks + 1) * BK;
      const int nb  = (buf ^ 1) * TILE_ELEMS;
      tdm_load_tile(&Xhi[gArow + k0n], (unsigned)(uintptr_t)&sAhi[nb]);
      tdm_load_tile(&Xlo[gArow + k0n], (unsigned)(uintptr_t)&sAlo[nb]);
      tdm_load_tile(&Yhi[gBrow + k0n], (unsigned)(uintptr_t)&sBhi[nb]);
      tdm_load_tile(&Ylo[gBrow + k0n], (unsigned)(uintptr_t)&sBlo[nb]);
    }

    const bf16_t* tAhi = sAhi + buf * TILE_ELEMS;
    const bf16_t* tAlo = sAlo + buf * TILE_ELEMS;
    const bf16_t* tBhi = sBhi + buf * TILE_ELEMS;
    const bf16_t* tBlo = sBlo + buf * TILE_ELEMS;

    // A fragment: 16-bit 16x32 layout => lane holds row (wave*16+l15),
    // K elems {8h..8h+7} U {16+8h..16+8h+7} -> two ds_load_b128.
    FragU ahi, alo;
    const int aOff = (wave * 16 + l15) * LDST;
    ahi.q[0] = *(const uint4*)&tAhi[aOff + 8 * h];
    ahi.q[1] = *(const uint4*)&tAhi[aOff + 16 + 8 * h];
    alo.q[0] = *(const uint4*)&tAlo[aOff + 8 * h];
    alo.q[1] = *(const uint4*)&tAlo[aOff + 16 + 8 * h];

#pragma unroll
    for (int nt = 0; nt < 8; ++nt) {
      // B = Y^T: lane holds Y-row (column of B) with identical K packing.
      FragU bhi, blo;
      const int bOff = (nt * 16 + l15) * LDST;
      bhi.q[0] = *(const uint4*)&tBhi[bOff + 8 * h];
      bhi.q[1] = *(const uint4*)&tBhi[bOff + 16 + 8 * h];
      blo.q[0] = *(const uint4*)&tBlo[bOff + 8 * h];
      blo.q[1] = *(const uint4*)&tBlo[bOff + 16 + 8 * h];
      // fp32-ish dot: hi*hi + hi*lo + lo*hi, f32 accumulate
      acc[nt] = __builtin_amdgcn_wmma_f32_16x16x32_bf16(false, ahi.v, false, bhi.v,
                                                        (short)0, acc[nt], false, false);
      acc[nt] = __builtin_amdgcn_wmma_f32_16x16x32_bf16(false, ahi.v, false, blo.v,
                                                        (short)0, acc[nt], false, false);
      acc[nt] = __builtin_amdgcn_wmma_f32_16x16x32_bf16(false, alo.v, false, bhi.v,
                                                        (short)0, acc[nt], false, false);
    }
    buf ^= 1;
  }

  // ---- Epilogue: sim = 1/(max(|x|^2+|y|^2-2*dot, 0)+1), fused reductions ----
  // C/D layout: VGPR v, lanes 0-15 -> (M = v + 8h, N = l15).
  const int rBase = rowBase + wave * 16 + 8 * h;
  const float4 nxa = *(const float4*)&nX[rBase];
  const float4 nxb = *(const float4*)&nX[rBase + 4];
  const float nxv[8] = {nxa.x, nxa.y, nxa.z, nxa.w, nxb.x, nxb.y, nxb.z, nxb.w};

  float rowAcc[8] = {};
#pragma unroll
  for (int nt = 0; nt < 8; ++nt) {
    const int gcol = colBase + nt * 16 + l15;
    const float ny = nY[gcol];
    float colpart = 0.f;
#pragma unroll
    for (int v = 0; v < 8; ++v) {
      float d2  = fmaxf(nxv[v] + ny - 2.f * acc[nt][v], 0.f);
      float sim = 1.f / (d2 + 1.f);
      if (WANT_DIAG || EXCL_DIAG) {
        if (rBase + v == gcol) {
          if (WANT_DIAG) diag[gcol] = sim;
          if (EXCL_DIAG) sim = 0.f;
        }
      }
      colpart += sim;
      // deterministic in-wave row-sum over the 16 columns of this tile
      float r = sim;
      r += __shfl_xor(r, 1);
      r += __shfl_xor(r, 2);
      r += __shfl_xor(r, 4);
      r += __shfl_xor(r, 8);
      rowAcc[v] += r;
    }
    if (WANT_COL) {
      colpart += __shfl_xor(colpart, 16);  // combine the two row-halves
      if (h == 0) sCol[wave * BN + nt * 16 + l15] = colpart;
    }
  }

  // unique (block, row) slot -> no atomics
  if (l15 == 0) {
    float* slot = rsPart + (size_t)blockIdx.x * NHALF;
#pragma unroll
    for (int v = 0; v < 8; ++v) slot[rBase + v] = rowAcc[v];
  }

  if (WANT_COL) {
    __syncthreads();
    if (tid < BN) {
      float s = 0.f;
#pragma unroll
      for (int w = 0; w < 8; ++w) s += sCol[w * BN + tid];  // fixed order
      csPart[(size_t)blockIdx.y * NHALF + colBase + tid] = s;
    }
  }
}

// ---------------------------------------------------------------------------
// 3) Fixed-order reduction of the 64 per-tile partial slots.
// ---------------------------------------------------------------------------
__global__ void reduce_partials(const float* __restrict__ p0, const float* __restrict__ p1,
                                const float* __restrict__ p2, const float* __restrict__ p3,
                                float* __restrict__ o0, float* __restrict__ o1,
                                float* __restrict__ o2, float* __restrict__ o3) {
  const int i = blockIdx.x * blockDim.x + threadIdx.x;  // < NHALF
  float s0 = 0.f, s1 = 0.f, s2 = 0.f, s3 = 0.f;
  for (int t = 0; t < TILES; ++t) {
    const size_t idx = (size_t)t * NHALF + i;
    s0 += p0[idx]; s1 += p1[idx]; s2 += p2[idx]; s3 += p3[idx];
  }
  o0[i] = s0; o1[i] = s1; o2[i] = s2; o3[i] = s3;
}

// ---------------------------------------------------------------------------
// 4) Final loss: logs, means, combine.
// ---------------------------------------------------------------------------
__global__ void final_loss_kernel(const float* __restrict__ diag,
                                  const float* __restrict__ rs_aa,
                                  const float* __restrict__ rs_bb,
                                  const float* __restrict__ rs_ab,
                                  const float* __restrict__ cs_ab,
                                  float* __restrict__ out) {
  __shared__ float red[256];
  __shared__ float res[3];
  const int t = threadIdx.x;
  float sA = 0.f, s1 = 0.f, s2 = 0.f;
  for (int i = t; i < NHALF; i += 256) {
    sA += logf(diag[i]);                 // tempered alignment
    s1 += logf(cs_ab[i] + rs_bb[i]);     // logsumexp_1 terms
    s2 += logf(rs_aa[i] + rs_ab[i]);     // logsumexp_2 terms
  }
  const float vals[3] = {sA, s1, s2};
  for (int k = 0; k < 3; ++k) {
    red[t] = vals[k];
    __syncthreads();
    for (int s = 128; s > 0; s >>= 1) {
      if (t < s) red[t] += red[t + s];
      __syncthreads();
    }
    if (t == 0) res[k] = red[0];
    __syncthreads();
  }
  if (t == 0) {
    const float n = (float)NHALF;
    const float alignment = res[0] / n;
    const float raw_uniformity = res[1] / n + res[2] / n;
    out[0] = -(alignment - 0.5f * raw_uniformity);
  }
}

// ---------------------------------------------------------------------------
extern "C" void kernel_launch(void* const* d_in, const int* in_sizes, int n_in,
                              void* d_out, int out_size, void* d_ws, size_t ws_size,
                              hipStream_t stream) {
  (void)in_sizes; (void)n_in; (void)out_size; (void)ws_size;
  const float* features = (const float*)d_in[0];   // [16384][256] f32
  float* out = (float*)d_out;                       // scalar f32

  // Workspace carve-up (~25.4 MB total)
  char* ws = (char*)d_ws;
  size_t off = 0;
  bf16_t* hi    = (bf16_t*)(ws + off); off += (size_t)NTOT * KDIM * sizeof(bf16_t);
  bf16_t* lo    = (bf16_t*)(ws + off); off += (size_t)NTOT * KDIM * sizeof(bf16_t);
  float*  norms = (float*)(ws + off);  off += (size_t)NTOT * sizeof(float);
  float* rsP_aa = (float*)(ws + off);  off += (size_t)TILES * NHALF * sizeof(float);
  float* rsP_bb = (float*)(ws + off);  off += (size_t)TILES * NHALF * sizeof(float);
  float* rsP_ab = (float*)(ws + off);  off += (size_t)TILES * NHALF * sizeof(float);
  float* csP_ab = (float*)(ws + off);  off += (size_t)TILES * NHALF * sizeof(float);
  float* diag   = (float*)(ws + off);  off += (size_t)NHALF * sizeof(float);
  float* rs_aa  = (float*)(ws + off);  off += (size_t)NHALF * sizeof(float);
  float* rs_bb  = (float*)(ws + off);  off += (size_t)NHALF * sizeof(float);
  float* rs_ab  = (float*)(ws + off);  off += (size_t)NHALF * sizeof(float);
  float* cs_ab  = (float*)(ws + off);  off += (size_t)NHALF * sizeof(float);

  prep_kernel<<<NTOT, KDIM, 0, stream>>>(features, hi, lo, norms);

  const bf16_t* ahi = hi;
  const bf16_t* alo = lo;
  const float*  nA  = norms;
  const bf16_t* bhi = hi + (size_t)NHALF * KDIM;
  const bf16_t* blo = lo + (size_t)NHALF * KDIM;
  const float*  nB  = norms + NHALF;

  const dim3 grid(TILES, TILES);
  // sim_aa: symmetric -> row sums only, diagonal zeroed
  sim_tile_kernel<false, false, true><<<grid, 256, 0, stream>>>(
      ahi, alo, nA, ahi, alo, nA, rsP_aa, csP_ab, diag);
  // sim_bb: symmetric -> row sums only, diagonal zeroed
  sim_tile_kernel<false, false, true><<<grid, 256, 0, stream>>>(
      bhi, blo, nB, bhi, blo, nB, rsP_bb, csP_ab, diag);
  // sim_ab: row sums + col sums + diagonal
  sim_tile_kernel<true, true, false><<<grid, 256, 0, stream>>>(
      ahi, alo, nA, bhi, blo, nB, rsP_ab, csP_ab, diag);

  reduce_partials<<<NHALF / 256, 256, 0, stream>>>(
      rsP_aa, rsP_bb, rsP_ab, csP_ab, rs_aa, rs_bb, rs_ab, cs_ab);

  final_loss_kernel<<<1, 256, 0, stream>>>(diag, rs_aa, rs_bb, rs_ab, cs_ab, out);
}
